// TopicRNN_58566174048819
// MI455X (gfx1250) — matrix operand order repaired and checked by reading, not compiled, not run on hardware
//
#include <hip/hip_runtime.h>
#include <hip/hip_bf16.h>

// Problem dims
constexpr int kV = 32000;
constexpr int kE = 512;
constexpr int kH = 1024;
constexpr int kB = 16;
constexpr int kS = 128;
constexpr int kK = 15;

typedef __attribute__((ext_vector_type(16))) __bf16 bf16x16;
typedef __attribute__((ext_vector_type(8)))  __bf16 bf16x8;
typedef __attribute__((ext_vector_type(8)))  float  f32x8;
typedef __attribute__((ext_vector_type(4)))  unsigned int u32x4;
typedef __attribute__((ext_vector_type(8)))  int i32x8;
typedef __attribute__((ext_vector_type(4)))  int i32x4;

__device__ __forceinline__ bf16x16 join8(bf16x8 lo, bf16x8 hi) {
  return __builtin_shufflevector(lo, hi, 0, 1, 2, 3, 4, 5, 6, 7,
                                         8, 9, 10, 11, 12, 13, 14, 15);
}

// A-fragment (16x32 bf16, M x K): lane r=lane&15 holds row r.
// sel=0 lanes: K = k0+0..7 and k0+16..23 ; sel=1: K = k0+8..15 and k0+24..31.
__device__ __forceinline__ bf16x16 load_a(const __bf16* src, int ld,
                                          int k0, int r, int sel) {
  const __bf16* p = src + (size_t)r * ld + k0 + sel * 8;
  bf16x8 lo = *(const bf16x8*)p;
  bf16x8 hi = *(const bf16x8*)(p + 16);
  return join8(lo, hi);
}

// B-fragment (32x16 bf16, K x N), B = W^T with W row-major [N, ld]:
// lane holds W row n = lane&15; K = k0+8*(lane>>4)+0..7 and k0+16+8*(lane>>4)+0..7.
__device__ __forceinline__ bf16x16 load_b(const __bf16* p) {
  bf16x8 lo = *(const bf16x8*)p;
  bf16x8 hi = *(const bf16x8*)(p + 16);
  return join8(lo, hi);
}

__device__ __forceinline__ f32x8 wmma_bf16(bf16x16 a, bf16x16 b, f32x8 c) {
  return __builtin_amdgcn_wmma_f32_16x16x32_bf16(false, a, false, b,
                                                 (short)0, c, false, false);
}

// ---------------------------------------------------------------- pre-passes
__global__ void cvt_f32_bf16(const float* __restrict__ in,
                             __bf16* __restrict__ out, int n) {
  int i = blockIdx.x * blockDim.x + threadIdx.x;
  int stride = gridDim.x * blockDim.x;
  for (; i < n; i += stride) out[i] = (__bf16)in[i];
}

__global__ void zero_u32(unsigned* p) { *p = 0u; }

__global__ void topic_gemv(const float* __restrict__ theta,
                           const float* __restrict__ beta,
                           float* __restrict__ ta) {
  int v = blockIdx.x * blockDim.x + threadIdx.x;
  if (v < kV) {
    float s = 0.f;
#pragma unroll
    for (int k = 0; k < kK; ++k) s += theta[k] * beta[k * kV + v];
    ta[v] = s;
  }
}

// Gather embeddings -> bf16, time-major X[s][b][e]
__global__ void embed_gather(const int* __restrict__ ids,
                             const float* __restrict__ Wemb,
                             __bf16* __restrict__ X) {
  int idx = blockIdx.x;            // 0 .. B*S-1
  int s = idx >> 4, b = idx & 15;
  int tok = ids[b * kS + s];
  const float* src = Wemb + (size_t)tok * kE;
  __bf16* dst = X + ((size_t)s * kB + b) * kE;
  for (int e = threadIdx.x; e < kE; e += blockDim.x) dst[e] = (__bf16)src[e];
}

// ---------------------------------------------------- parallel input project
// preact[(s,b), n] = X[(s,b), :] . W_ih[n, :] + b_ih[n] + b_hh[n]
__global__ void __launch_bounds__(256)
preact_gemm(const __bf16* __restrict__ X, const __bf16* __restrict__ Wih,
            const float* __restrict__ b_ih, const float* __restrict__ b_hh,
            float* __restrict__ preact) {
  int wave = threadIdx.x >> 5, lane = threadIdx.x & 31;
  int r = lane & 15, sel = lane >> 4;
  int srow = blockIdx.x;                    // M-tile = (s fixed, b=0..15)
  const __bf16* As = X + (size_t)srow * kB * kE;
  int tile0 = wave * 8;                     // 64 N-tiles over 8 waves

  f32x8 acc[8] = {};
  for (int k = 0; k < kE; k += 32) {
    bf16x16 a = load_a(As, kE, k, r, sel);
#pragma unroll
    for (int t = 0; t < 8; ++t) {
      bf16x16 bw = load_b(Wih + (size_t)((tile0 + t) * 16 + r) * kE + k + sel * 8);
      acc[t] = wmma_bf16(a, bw, acc[t]);
    }
  }
  float* Crow = preact + (size_t)srow * kB * kH;
#pragma unroll
  for (int t = 0; t < 8; ++t) {
    int n = (tile0 + t) * 16 + r;
    float bias = b_ih[n] + b_hh[n];
#pragma unroll
    for (int j = 0; j < 8; ++j) {
      int m = sel * 8 + j;
      Crow[(size_t)m * kH + n] = acc[t][j] + bias;
    }
  }
}

// ------------------------------------------------------------------- RNN scan
// 8 cooperating workgroups (ideally one workgroup cluster), each owning 128
// columns of h. Per step each WG streams only its 256KB W_hh slice from L2.
// h exchanged via double-buffered global buffer; one cross-WG barrier/step:
// CDNA5 cluster barrier fast path + device-scope atomic counter guarantee.
constexpr int kRG = 8;   // cooperating workgroups
__global__ void __launch_bounds__(256)
rnn_scan(const float* __restrict__ preact, const __bf16* __restrict__ Whh,
         const float* __restrict__ h0, __bf16* __restrict__ outs,
         float* __restrict__ hT, __bf16* __restrict__ hx,
         unsigned* __restrict__ cnt) {
  __shared__ __bf16 hbuf[kB][kH];   // 32 KB: full h_t, staged each step
  int tid = threadIdx.x;
  int wg = blockIdx.x;              // 0..kRG-1
  int wave = tid >> 5, lane = tid & 31;
  int r = lane & 15, sel = lane >> 4;
  int n0 = wg * (kH / kRG) + wave * 16;   // this wave's 16 columns

  for (int i = tid; i < kB * kH; i += blockDim.x)
    hbuf[0][i] = (__bf16)h0[i];
  __syncthreads();

  for (int t = 0; t < kS; ++t) {
    // Launder weight pointer: stops LICM from hoisting all B-fragments out of
    // the time loop (which spilled to scratch); weights re-stream from L2.
    const __bf16* Wt = Whh;
    asm volatile("" : "+s"(Wt));

    const float* P = preact + (size_t)t * kB * kH;
    f32x8 acc;
#pragma unroll
    for (int j = 0; j < 8; ++j)
      acc[j] = P[(size_t)(sel * 8 + j) * kH + n0 + r];

    // h @ W_hh^T slice: A from LDS (ds_load_b128), B streamed from L2
    for (int k = 0; k < kH; k += 32) {
      bf16x16 a  = load_a(&hbuf[0][0], kH, k, r, sel);
      bf16x16 bw = load_b(Wt + (size_t)(n0 + r) * kH + k + sel * 8);
      acc = wmma_bf16(a, bw, acc);
    }

    __bf16* hxw = hx + ((t + 1) & 1) * (kB * kH);   // h_{t+1} buffer
    __bf16* outrow = outs + (size_t)t * kB * kH;
#pragma unroll
    for (int j = 0; j < 8; ++j) {
      int m = sel * 8 + j;          // C layout: M = 8*sel + j, N = lane&15
      float v = tanhf(acc[j]);
      __bf16 q = (__bf16)v;
      hxw[(size_t)m * kH + n0 + r] = q;
      outrow[(size_t)m * kH + n0 + r] = q;
      if (t == kS - 1) hT[(size_t)m * kH + n0 + r] = v;
    }

    // ---------------- cross-workgroup step barrier ----------------
    __threadfence();                 // release slice stores to device scope
    __syncthreads();                 // whole WG done writing
    if (tid == 0)
      __hip_atomic_fetch_add(cnt, 1u, __ATOMIC_RELEASE,
                             __HIP_MEMORY_SCOPE_AGENT);
    // Cluster barrier: one wave signals, all waves wait. Hardware-speed sync
    // when launched as a cluster; NOP otherwise (atomic barrier below is the
    // portable guarantee either way).
    if (wave == 0) asm volatile("s_barrier_signal -3" ::: "memory");
    asm volatile("s_barrier_wait -3" ::: "memory");
    if (tid == 0) {
      unsigned need = (unsigned)(kRG * (t + 1));
      while (__hip_atomic_load(cnt, __ATOMIC_ACQUIRE,
                               __HIP_MEMORY_SCOPE_AGENT) < need)
        __builtin_amdgcn_s_sleep(1);
    }
    __syncthreads();                 // acquire (incl. WGP$ inv) ordered before
                                     // everyone's staging loads

    if (t < kS - 1) {
      // stage full h_{t+1} (written by all 8 WGs) into LDS for next step
      const __bf16* src = hxw;
      for (int i = tid * 8; i < kB * kH; i += blockDim.x * 8)
        *(bf16x8*)&hbuf[0][i] = *(const bf16x8*)&src[i];
      __syncthreads();
    }
  }
}

// -------------------------------------------------------------- decoder GEMM
// C[(b,s), v] = outs[s][b][:] . W_dec[v][:] + b_dec[v] + topic[v]*(stop==0)
// Each block: M=64 (4 s-tiles, 128KB contiguous A panel staged to LDS via one
// TDM descriptor) x N=512 (8 waves x 4 N-tiles). Every B-fragment from L2
// feeds 4 WMMAs -> 64 FLOP per L2 byte, 4x less W_dec re-read.
constexpr int kMT = 4;     // M-tiles (s values) per block
constexpr int kNT = 4;     // N-tiles per wave
__global__ void __launch_bounds__(256)
decoder_gemm(const __bf16* __restrict__ A, const __bf16* __restrict__ Wd,
             const float* __restrict__ b_dec, const float* __restrict__ ta,
             const int* __restrict__ stop, float* __restrict__ out) {
  __shared__ __bf16 Ash[kMT * kB * kH];   // 128 KB A panel (2 blocks/WGP)
  int wave = threadIdx.x >> 5, lane = threadIdx.x & 31;
  int r = lane & 15, sel = lane >> 4;
  int s0 = blockIdx.x * kMT;
  int tile0 = (blockIdx.y * 8 + wave) * kNT;
  const __bf16* As = A + (size_t)s0 * kB * kH;   // 64 contiguous rows

#if __has_builtin(__builtin_amdgcn_tensor_load_to_lds) && \
    __has_builtin(__builtin_amdgcn_s_wait_tensorcnt)
  if (threadIdx.x == 0) {
    // 1D DMA: 131072 B as 16384 x 8-byte elements (data_size=3).
    unsigned long long ga = (unsigned long long)(const void*)As;
    unsigned lds = (unsigned)(unsigned long long)(const void*)&Ash[0];
    u32x4 g0 = { 1u,                                   // count=1, no gather
                 lds,                                  // lds_addr
                 (unsigned)ga,                         // global_addr[31:0]
                 (unsigned)((ga >> 32) & 0x1FFFFFFu) | (2u << 30) }; // type=2
    i32x8 g1 = { (int)(3u << 16),           // data_size = 8B
                 (int)(16384u << 16),       // tensor_dim0[15:0]
                 (int)(1u << 16),           // tensor_dim0[31:16]=0, tensor_dim1=1
                 (int)(16384u << 16),       // tile_dim0 = 16384
                 1,                         // tile_dim1 = 1, tile_dim2 = 0
                 16384,                     // tensor_dim0_stride
                 0, 0 };
    i32x4 z4 = { 0, 0, 0, 0 };
#if __clang_major__ >= 23
    i32x8 z8 = { 0, 0, 0, 0, 0, 0, 0, 0 };
    __builtin_amdgcn_tensor_load_to_lds(g0, g1, z4, z4, z8, 0);
#else
    __builtin_amdgcn_tensor_load_to_lds(g0, g1, z4, z4, 0);
#endif
  }
  __builtin_amdgcn_s_wait_tensorcnt(0);
  __syncthreads();
#else
  for (int i = threadIdx.x * 8; i < kMT * kB * kH; i += blockDim.x * 8)
    *(bf16x8*)&Ash[i] = *(const bf16x8*)&As[i];
  __syncthreads();
#endif

  f32x8 acc[kMT][kNT] = {};
  for (int k = 0; k < kH; k += 32) {
    bf16x16 a[kMT];
#pragma unroll
    for (int mt = 0; mt < kMT; ++mt)
      a[mt] = load_a(&Ash[mt * kB * kH], kH, k, r, sel);   // ds_load_b128
#pragma unroll
    for (int nt = 0; nt < kNT; ++nt) {
      int nr = (tile0 + nt) * 16 + r;
      if (nr > kV - 1) nr = kV - 1;          // clamp OOB tiles (stores guarded)
      bf16x16 bw = load_b(Wd + (size_t)nr * kH + k + sel * 8);
#pragma unroll
      for (int mt = 0; mt < kMT; ++mt)
        acc[mt][nt] = wmma_bf16(a[mt], bw, acc[mt][nt]);
    }
  }

#pragma unroll
  for (int mt = 0; mt < kMT; ++mt) {
    int s = s0 + mt;
    float mask[8];
#pragma unroll
    for (int j = 0; j < 8; ++j) {
      int b = sel * 8 + j;
      mask[j] = (stop[b * kS + s] == 0) ? 1.f : 0.f;
    }
#pragma unroll
    for (int nt = 0; nt < kNT; ++nt) {
      int nc0 = (tile0 + nt) * 16;
      if (nc0 < kV) {
        int n = nc0 + r;
        float bias = b_dec[n];
        float tadd = ta[n];
#pragma unroll
        for (int j = 0; j < 8; ++j) {
          int b = sel * 8 + j;
          out[((size_t)b * kS + s) * kV + n] =
              acc[mt][nt][j] + bias + tadd * mask[j];
        }
      }
    }
  }
}

// -------------------------------------------------------------------- launch
extern "C" void kernel_launch(void* const* d_in, const int* in_sizes, int n_in,
                              void* d_out, int out_size, void* d_ws, size_t ws_size,
                              hipStream_t stream) {
  const int*   input_ids = (const int*)  d_in[0];
  const float* hidden    = (const float*)d_in[1];
  const int*   stop      = (const int*)  d_in[2];
  const float* W_emb     = (const float*)d_in[3];
  const float* W_ih      = (const float*)d_in[4];
  const float* b_ih      = (const float*)d_in[5];
  const float* W_hh      = (const float*)d_in[6];
  const float* b_hh      = (const float*)d_in[7];
  const float* W_dec     = (const float*)d_in[8];
  const float* b_dec     = (const float*)d_in[9];
  const float* beta      = (const float*)d_in[10];
  const float* theta     = (const float*)d_in[11];

  char* ws = (char*)d_ws;
  size_t off = 0;
  auto carve = [&](size_t bytes) -> char* {
    char* p = ws + off;
    off += (bytes + 255) & ~(size_t)255;
    return p;
  };
  __bf16*   Wdec_bf = (__bf16*)carve((size_t)kV * kH * 2);        // 65.5 MB
  __bf16*   Whh_bf  = (__bf16*)carve((size_t)kH * kH * 2);        // 2 MB
  __bf16*   Wih_bf  = (__bf16*)carve((size_t)kH * kE * 2);        // 1 MB
  __bf16*   X_bf    = (__bf16*)carve((size_t)kS * kB * kE * 2);   // 2 MB
  __bf16*   outs_bf = (__bf16*)carve((size_t)kS * kB * kH * 2);   // 4 MB
  float*    preact  = (float*) carve((size_t)kS * kB * kH * 4);   // 8 MB
  float*    topic   = (float*) carve((size_t)kV * 4);             // 128 KB
  __bf16*   hx      = (__bf16*)carve((size_t)2 * kB * kH * 2);    // 64 KB
  unsigned* cnt     = (unsigned*)carve(256);

  float* out = (float*)d_out;                       // decoded [B,S,V]
  float* hT  = out + (size_t)kB * kS * kV;          // then hT [1,B,H]

  // bf16 conversion pre-passes (memory-bound streams)
  cvt_f32_bf16<<<2048, 256, 0, stream>>>(W_dec, Wdec_bf, kV * kH);
  cvt_f32_bf16<<<512,  256, 0, stream>>>(W_hh,  Whh_bf,  kH * kH);
  cvt_f32_bf16<<<256,  256, 0, stream>>>(W_ih,  Wih_bf,  kH * kE);
  zero_u32    <<<1, 1, 0, stream>>>(cnt);
  topic_gemv  <<<(kV + 255) / 256, 256, 0, stream>>>(theta, beta, topic);
  embed_gather<<<kB * kS, 128, 0, stream>>>(input_ids, W_emb, X_bf);

  // parallel input projection (x @ W_ih^T + biases) for all timesteps
  preact_gemm<<<kS, 256, 0, stream>>>(X_bf, Wih_bf, b_ih, b_hh, preact);

  // sequential RNN scan across 8 cooperating workgroups
  rnn_scan<<<kRG, 256, 0, stream>>>(preact, Whh_bf, hidden, outs_bf, hT,
                                    hx, cnt);

  // big vocab projection: M=64 per block, N=512 per block
  int ntiles = kV / 16;                                       // 2000
  dim3 dgrid(kS / kMT, (ntiles + 8 * kNT - 1) / (8 * kNT));   // (32, 63)
  decoder_gemm<<<dgrid, 256, 0, stream>>>(outs_bf, Wdec_bf, b_dec, topic,
                                          stop, out);
}
